// Attention_712964571194
// MI455X (gfx1250) — matrix operand order repaired and checked by reading, not compiled, not run on hardware
//
#include <hip/hip_runtime.h>

typedef __attribute__((ext_vector_type(16))) __bf16 v16bf;
typedef __attribute__((ext_vector_type(8)))  __bf16 v8bf;
typedef __attribute__((ext_vector_type(8)))  float  v8f;
typedef __attribute__((ext_vector_type(4)))  int    v4i;

#define DEV __device__ __forceinline__

#if defined(__gfx1250__) && __has_builtin(__builtin_amdgcn_global_load_async_to_lds_b128) && \
    __has_builtin(__builtin_amdgcn_s_wait_asynccnt)
#define USE_ASYNC_COPY 1
#else
#define USE_ASYNC_COPY 0
#endif

#if USE_ASYNC_COPY
// Direct global -> LDS 16B copy, no VGPR bounce; tracked by ASYNCcnt.
// Builtin signature (from hipcc diagnostic): (v4i AS1*, v4i AS3*, imm offset, imm cpol)
DEV void async_copy16(const void* g, void* l) {
  __builtin_amdgcn_global_load_async_to_lds_b128(
      (__attribute__((address_space(1))) v4i*)g,
      (__attribute__((address_space(3))) v4i*)l, 0, 0);
}
DEV void async_wait0() { __builtin_amdgcn_s_wait_asynccnt(0); }
#endif

DEV v16bf cat16(v8bf a, v8bf b) {
  return __builtin_shufflevector(a, b, 0,1,2,3,4,5,6,7,8,9,10,11,12,13,14,15);
}
DEV v8bf cvt8(float4 a, float4 b) {
  v8bf r;
  r[0]=(__bf16)a.x; r[1]=(__bf16)a.y; r[2]=(__bf16)a.z; r[3]=(__bf16)a.w;
  r[4]=(__bf16)b.x; r[5]=(__bf16)b.y; r[6]=(__bf16)b.z; r[7]=(__bf16)b.w;
  return r;
}
DEV v8f wmma_bf16(v16bf a, v16bf b, v8f c) {
  return __builtin_amdgcn_wmma_f32_16x16x32_bf16(false, a, false, b, (short)0, c, false, false);
}

// ---------------------------------------------------------------------------
// Weight transpose + f32 -> bf16 convert:  W (512 k x 512 n) -> Wt (n x k)
// ---------------------------------------------------------------------------
__global__ __launch_bounds__(256) void transpose_w_bf16(
    const float* __restrict__ W, __bf16* __restrict__ Wt) {
  const int idx = blockIdx.x * 256 + threadIdx.x;    // 512*512 total
  const int n = idx >> 9, k = idx & 511;
  Wt[idx] = (__bf16)W[(size_t)k * 512 + n];
}

// ---------------------------------------------------------------------------
// WMMA GEMM:  C(M x 512) = A(M x 512) * Bt^T   (Bt stored n-major: 512n x 512k)
//   Block tile 128x64, 8 waves, each wave: 16 rows x 64 cols (4 C tiles).
//   MODE 0: out bf16 (B,H,S,64)    [Q (alpha=0.125) and K]
//   MODE 1: out bf16 (B,H,64,S)    [V transposed]
//   MODE 2: out f32 row-major + bias  [final projection -> d_out]
// ---------------------------------------------------------------------------
template<int MODE, bool ABF16>
__global__ __launch_bounds__(256) void gemm512(
    const void* __restrict__ Aptr, const __bf16* __restrict__ Bt,
    const float* __restrict__ bias, void* __restrict__ Out, float alphaScale) {
  constexpr int Kd = 512;
  __shared__ __align__(16) __bf16 lA[128 * 40];   // 128 rows x 32 k, stride 40 halves
  __shared__ __align__(16) __bf16 lB[64 * 40];    // 64 n-rows x 32 k

  const int tid  = threadIdx.x;
  const int wave = tid >> 5, lane = tid & 31;
  const int lm   = lane & 15, lhi = lane >> 4;
  const int m0 = blockIdx.x * 128;
  const int n0 = blockIdx.y * 64;

  const int arow = (tid * 16) >> 5;      // 0..127, 16 elems (half row) per thread
  const int acol = (tid * 16) & 31;      // 0 or 16
  const int brow = (tid * 8) >> 5;       // 0..63, 8 halves per thread
  const int bcol = (tid * 8) & 31;

  v8f acc[4] = {};

  for (int k0 = 0; k0 < Kd; k0 += 32) {
    __syncthreads();
    if (ABF16) {
      const __bf16* A = (const __bf16*)Aptr + (size_t)(m0 + arow) * Kd + k0 + acol;
#if USE_ASYNC_COPY
      async_copy16(A,     &lA[arow * 40 + acol]);
      async_copy16(A + 8, &lA[arow * 40 + acol + 8]);
#else
      uint4 d0 = ((const uint4*)A)[0];
      uint4 d1 = ((const uint4*)A)[1];
      *(uint4*)&lA[arow * 40 + acol]     = d0;
      *(uint4*)&lA[arow * 40 + acol + 8] = d1;
#endif
    } else {
      const float* A = (const float*)Aptr + (size_t)(m0 + arow) * Kd + k0 + acol;
      float4 f0 = ((const float4*)A)[0];
      float4 f1 = ((const float4*)A)[1];
      float4 f2 = ((const float4*)A)[2];
      float4 f3 = ((const float4*)A)[3];
      *(v8bf*)&lA[arow * 40 + acol]     = cvt8(f0, f1);
      *(v8bf*)&lA[arow * 40 + acol + 8] = cvt8(f2, f3);
    }
    {
      const __bf16* B = Bt + (size_t)(n0 + brow) * Kd + k0 + bcol;
#if USE_ASYNC_COPY
      async_copy16(B, &lB[brow * 40 + bcol]);
#else
      *(uint4*)&lB[brow * 40 + bcol] = *(const uint4*)B;
#endif
    }
    if (k0 + 32 < Kd) {
      if (ABF16)
        __builtin_prefetch((const __bf16*)Aptr + (size_t)(m0 + arow) * Kd + k0 + 32 + acol, 0, 1);
      else
        __builtin_prefetch((const float*)Aptr + (size_t)(m0 + arow) * Kd + k0 + 32 + acol, 0, 1);
    }
#if USE_ASYNC_COPY
    async_wait0();
#endif
    __syncthreads();

    const int ar = wave * 16 + lm;
    // A fragment 16x32: halves 0-7 -> k = lhi*8 + 0..7 ; halves 8-15 -> k = 16 + lhi*8 + 0..7
    v16bf afrag = cat16(*(const v8bf*)&lA[ar * 40 + lhi * 8],
                        *(const v8bf*)&lA[ar * 40 + 16 + lhi * 8]);
#pragma unroll
    for (int c = 0; c < 4; ++c) {
      // B fragment 32x16: lane column n = 16c+lm; halves 0-7 -> k = lhi*8 + 0..7 ; halves 8-15 -> k = 16 + lhi*8 + 0..7
      v16bf bfrag = cat16(*(const v8bf*)&lB[(16 * c + lm) * 40 + lhi * 8],
                          *(const v8bf*)&lB[(16 * c + lm) * 40 + 16 + lhi * 8]);
      acc[c] = wmma_bf16(afrag, bfrag, acc[c]);
    }
  }

#pragma unroll
  for (int c = 0; c < 4; ++c) {
    const int n = n0 + 16 * c + lm;
    if (MODE == 0) {
      __bf16* O = (__bf16*)Out;
      const int h = n >> 6, d = n & 63;
#pragma unroll
      for (int r = 0; r < 8; ++r) {
        const int m = m0 + wave * 16 + 8 * lhi + r;
        const int b = m >> 13, s = m & 8191;
        O[(((size_t)b * 8 + h) * 8192 + s) * 64 + d] = (__bf16)(acc[c][r] * alphaScale);
      }
    } else if (MODE == 1) {
      const int h = n >> 6, d = n & 63;
      const int mb = m0 + wave * 16 + 8 * lhi;
      const int b = mb >> 13, s = mb & 8191;
      v8bf pk;
#pragma unroll
      for (int r = 0; r < 8; ++r) pk[r] = (__bf16)(acc[c][r] * alphaScale);
      *(v8bf*)((__bf16*)Out + (((size_t)b * 8 + h) * 64 + d) * 8192 + s) = pk;
    } else {
      float* O = (float*)Out;
      const float bv = bias[n];
#pragma unroll
      for (int r = 0; r < 8; ++r) {
        const int m = m0 + wave * 16 + 8 * lhi + r;
        O[(size_t)m * 512 + n] = acc[c][r] + bv;
      }
    }
  }
}

// ---------------------------------------------------------------------------
// Flash attention, one (batch*head, 128-query block) per WG; 8 waves x 16 q.
// Computes S^T = K x Q^T so exp(S)->A-operand repack is purely per-lane.
//   Q,K: (16, 8192, 64) bf16 (Q pre-scaled by 1/sqrt(64));  Vt: (16, 64, 8192)
//   Out: attn bf16 (B*S, 512) row-major.
// ---------------------------------------------------------------------------
__global__ __launch_bounds__(256) void flash_attn(
    const __bf16* __restrict__ Q, const __bf16* __restrict__ Kmat,
    const __bf16* __restrict__ Vt, __bf16* __restrict__ Oout) {
  constexpr int S = 8192, DH = 64;
  __shared__ __align__(16) __bf16 lK[32 * 72];   // 32 keys x 64 d, stride 72 halves
  __shared__ __align__(16) __bf16 lV[64 * 40];   // 64 d x 32 keys, stride 40 halves

  const int tid  = threadIdx.x;
  const int wave = tid >> 5, lane = tid & 31;
  const int lm   = lane & 15, lhi = lane >> 4;
  const int bh = blockIdx.x;            // b*8 + h
  const int q0 = blockIdx.y * 128;

  // Q as B operand of K x Q^T: lane holds query q0+wave*16+lm, d contiguous.
  const __bf16* qp = Q + ((size_t)bh * S + (q0 + wave * 16 + lm)) * DH;
  const v16bf bq0 = cat16(*(const v8bf*)(qp + lhi * 16),
                          *(const v8bf*)(qp + lhi * 16 + 8));
  const v16bf bq1 = cat16(*(const v8bf*)(qp + 32 + lhi * 16),
                          *(const v8bf*)(qp + 32 + lhi * 16 + 8));

  const __bf16* Kb = Kmat + (size_t)bh * S * DH;
  const __bf16* Vb = Vt + (size_t)bh * DH * S;

  const int krow = (tid * 8) >> 6;      // 0..31
  const int kcol = (tid * 8) & 63;
  const int vrow = tid >> 2;            // 0..63
  const int vcol = (tid & 3) * 8;

  float m_run = -3.0e38f, l_run = 0.f;
  v8f o[4] = {};

  for (int k0 = 0; k0 < S; k0 += 32) {
    __syncthreads();
#if USE_ASYNC_COPY
    async_copy16(Kb + (size_t)(k0 + krow) * DH + kcol, &lK[krow * 72 + kcol]);
    async_copy16(Vb + (size_t)vrow * S + k0 + vcol,    &lV[vrow * 40 + vcol]);
#else
    *(uint4*)&lK[krow * 72 + kcol] = *(const uint4*)(Kb + (size_t)(k0 + krow) * DH + kcol);
    *(uint4*)&lV[vrow * 40 + vcol] = *(const uint4*)(Vb + (size_t)vrow * S + k0 + vcol);
#endif
    if (k0 + 32 < S) {
      __builtin_prefetch(Kb + (size_t)(k0 + 32 + krow) * DH + kcol, 0, 1);
      __builtin_prefetch(Vb + (size_t)vrow * S + k0 + 32 + vcol, 0, 1);
    }
#if USE_ASYNC_COPY
    async_wait0();
#endif
    __syncthreads();

    // S^T tiles: two 16key x 16query C tiles, reducing over d = 64 in 2 steps.
    v8f st[2];
#pragma unroll
    for (int t = 0; t < 2; ++t) {
      const int row = t * 16 + lm;      // key row within the 32-key tile
      v16bf a0 = cat16(*(const v8bf*)&lK[row * 72 + lhi * 16],
                       *(const v8bf*)&lK[row * 72 + lhi * 16 + 8]);
      v16bf a1 = cat16(*(const v8bf*)&lK[row * 72 + 32 + lhi * 16],
                       *(const v8bf*)&lK[row * 72 + 32 + lhi * 16 + 8]);
      v8f c = {};
      c = wmma_bf16(a0, bq0, c);
      c = wmma_bf16(a1, bq1, c);
      st[t] = c;
    }

    // Online softmax: lane holds 16 of the 32 scores for query lm; partner lane
    // (lane^16) holds the rest -> one shfl_xor combines row stats.
    float mloc = -3.0e38f;
#pragma unroll
    for (int t = 0; t < 2; ++t)
#pragma unroll
      for (int r = 0; r < 8; ++r) mloc = fmaxf(mloc, st[t][r]);
    mloc = fmaxf(mloc, __shfl_xor(mloc, 16, 32));
    const float m_new = fmaxf(m_run, mloc);

    float psum = 0.f;
    v16bf p;    // A-operand of P x V: half j = t*8+r (per-lane repack, no shuffle)
#pragma unroll
    for (int t = 0; t < 2; ++t)
#pragma unroll
      for (int r = 0; r < 8; ++r) {
        const float e = __expf(st[t][r] - m_new);
        psum += e;
        p[t * 8 + r] = (__bf16)e;
      }
    psum += __shfl_xor(psum, 16, 32);
    const float resc = __expf(m_run - m_new);
    l_run = l_run * resc + psum;
    m_run = m_new;

    // Rescale O accumulators: O-tile VGPR r holds query 8*lhi + r.
    float rs[8];
#pragma unroll
    for (int r = 0; r < 8; ++r) rs[r] = __shfl(resc, r + lhi * 24, 32);
#pragma unroll
    for (int c = 0; c < 4; ++c)
#pragma unroll
      for (int r = 0; r < 8; ++r) o[c][r] *= rs[r];

    // O += P(16q x 32k) x V(32k x 16d) for 4 d-column tiles.
#pragma unroll
    for (int c = 0; c < 4; ++c) {
      const int vr = 16 * c + lm;
      v16bf bv = cat16(*(const v8bf*)&lV[vr * 40 + lhi * 8],
                       *(const v8bf*)&lV[vr * 40 + 16 + lhi * 8]);
      o[c] = wmma_bf16(p, bv, o[c]);
    }
  }

  float ls[8];
#pragma unroll
  for (int r = 0; r < 8; ++r) ls[r] = __shfl(l_run, r + lhi * 24, 32);

  const int b = bh >> 3, h = bh & 7;
#pragma unroll
  for (int c = 0; c < 4; ++c) {
    const int d = 16 * c + lm;
#pragma unroll
    for (int r = 0; r < 8; ++r) {
      const int s = q0 + wave * 16 + 8 * lhi + r;
      Oout[((size_t)b * S + s) * 512 + h * 64 + d] = (__bf16)(o[c][r] / ls[r]);
    }
  }
}

// ---------------------------------------------------------------------------
extern "C" void kernel_launch(void* const* d_in, const int* in_sizes, int n_in,
                              void* d_out, int out_size, void* d_ws, size_t ws_size,
                              hipStream_t stream) {
  (void)in_sizes; (void)n_in; (void)out_size; (void)ws_size;
  const float* X  = (const float*)d_in[0];   // (2, 8192, 512)
  const float* wq = (const float*)d_in[1];
  const float* wk = (const float*)d_in[2];
  const float* wv = (const float*)d_in[3];
  const float* wo = (const float*)d_in[4];
  const float* bo = (const float*)d_in[5];

  char* ws = (char*)d_ws;
  size_t off = 0;
  auto take = [&](size_t elems) {
    __bf16* p = (__bf16*)(ws + off);
    off += elems * sizeof(__bf16);
    return p;
  };
  __bf16* wtq  = take((size_t)512 * 512);
  __bf16* wtk  = take((size_t)512 * 512);
  __bf16* wtv  = take((size_t)512 * 512);
  __bf16* wto  = take((size_t)512 * 512);
  __bf16* Qb   = take((size_t)16 * 8192 * 64);
  __bf16* Kb   = take((size_t)16 * 8192 * 64);
  __bf16* Vtb  = take((size_t)16 * 64 * 8192);
  __bf16* attn = take((size_t)16384 * 512);

  transpose_w_bf16<<<1024, 256, 0, stream>>>(wq, wtq);
  transpose_w_bf16<<<1024, 256, 0, stream>>>(wk, wtk);
  transpose_w_bf16<<<1024, 256, 0, stream>>>(wv, wtv);
  transpose_w_bf16<<<1024, 256, 0, stream>>>(wo, wto);

  dim3 gg(128, 8);  // M/128 x N/64
  gemm512<0, false><<<gg, 256, 0, stream>>>(X, wtq, nullptr, Qb, 0.125f);  // Q * 1/sqrt(64)
  gemm512<0, false><<<gg, 256, 0, stream>>>(X, wtk, nullptr, Kb, 1.0f);
  gemm512<1, false><<<gg, 256, 0, stream>>>(X, wtv, nullptr, Vtb, 1.0f);   // V transposed

  flash_attn<<<dim3(16, 64), 256, 0, stream>>>(Qb, Kb, Vtb, attn);

  gemm512<2, true><<<gg, 256, 0, stream>>>(attn, wto, bo, d_out, 1.0f);
}
